// CSPN_3D_58660663328882
// MI455X (gfx1250) — compile-verified, matrix operand order checked by reading, not run on hardware
//
#include <hip/hip_runtime.h>
#include <hip/hip_bf16.h>

typedef __attribute__((ext_vector_type(16))) __bf16 v16bf;
typedef __attribute__((ext_vector_type(8)))  float  v8f;

#define B_   2
#define CH_  32
#define D_   32
#define H_   64
#define W_   128
#define NP   (B_*D_*H_*W_)      // 524288 voxels
#define NTAP 27
#define KDIM (NTAP*CH_)         // 864, ordered tap-major: k = t*32 + c

// workspace layout (bytes)
#define WS_WMAT 0                              // bf16 [32][864]  = 55296 B
#define WS_WPL  65536                          // bf16 27 planes of NP
#define WS_X0   (WS_WPL + 27*(size_t)NP*2)     // f32 NP
#define WS_X1   (WS_X0 + (size_t)NP*4)         // f32 NP

// LDS geometry for conv: patch [3z][4y][130x][32c] bf16 + Wmat [32][864] bf16
#define PATCH_ELEMS (3*4*130*CH_)              // 49920 elems = 99840 B
#define PATCH_BYTES (PATCH_ELEMS*2)
#define ROWSTRIDE   (130*CH_)                  // 4160 elems per (z,y) row

// ---------------------------------------------------------------------------
// Repack conv weights into bf16 B-matrix Wmat[n][k], k = tap*32 + channel.
// n: 0..25 = w_g out-channels, 26 = w_c, 27..31 = zero padding.
// ---------------------------------------------------------------------------
__global__ __launch_bounds__(256) void prep_weights(const float* __restrict__ wg,
                                                    const float* __restrict__ wc,
                                                    __bf16* __restrict__ wmat) {
    for (int i = blockIdx.x * 256 + threadIdx.x; i < 32 * KDIM;
         i += gridDim.x * 256) {
        int n = i / KDIM, k = i % KDIM;
        int t = k >> 5, c = k & 31;
        float v = 0.f;
        if (n < 26)       v = wg[(n * CH_ + c) * NTAP + t];
        else if (n == 26) v = wc[c * NTAP + t];
        wmat[(size_t)n * KDIM + k] = (__bf16)v;
    }
}

// ---------------------------------------------------------------------------
// Implicit-GEMM 3x3x3 conv via WMMA bf16 + fused CSPN weight normalization.
// One block = two (b,z,y) rows = 256 voxels. 8 waves; wave wv owns output row
// r = wv>>2 and two 16-voxel M-tiles at xbase=(wv&3)*32, sharing B fragments:
// per tap 8x ds_load_b128 feed 4 WMMAs (2 DS ops / WMMA).
// ---------------------------------------------------------------------------
__global__ __launch_bounds__(256) void conv_wmma(const float* __restrict__ xin,
                                                 const __bf16* __restrict__ wmat,
                                                 __bf16* __restrict__ wpl,
                                                 float* __restrict__ x0) {
    __shared__ __align__(16) char smem[PATCH_BYTES + 32 * KDIM * 2];
    __bf16* patch  = (__bf16*)smem;
    __bf16* wlds   = (__bf16*)(smem + PATCH_BYTES);
    float*  cstage = (float*)smem;     // 256*32*4 = 32 KB, aliases patch later

    const int tid   = threadIdx.x;
    const int bid   = blockIdx.x;                 // (b*D + z)*(H/2) + ypair
    const int ybase = (bid % (H_ / 2)) * 2;
    const int zc    = (bid / (H_ / 2)) % D_;
    const int b     = bid / ((H_ / 2) * D_);

    // stage weight matrix (global, L2-hot) -> LDS as dwords
    {
        const unsigned int* src = (const unsigned int*)wmat;
        unsigned int*       dst = (unsigned int*)wlds;
        for (int i = tid; i < (32 * KDIM) / 2; i += 256) dst[i] = src[i];
    }
    // stage zero-padded input patch, f32 -> bf16, channel-innermost
    for (int i = tid; i < PATCH_ELEMS; i += 256) {
        int x0i = i % 130; int r2 = i / 130;
        int y0 = r2 % 4; r2 /= 4;
        int z0 = r2 % 3; int c = r2 / 3;
        int gz = zc + z0 - 1, gy = ybase + y0 - 1, gx = x0i - 1;
        float v = 0.f;
        if ((unsigned)gz < D_ && (unsigned)gy < H_ && (unsigned)gx < W_)
            v = xin[((((size_t)b * CH_ + c) * D_ + gz) * H_ + gy) * W_ + gx];
        patch[((z0 * 4 + y0) * 130 + x0i) * CH_ + c] = (__bf16)v;
    }
    __syncthreads();

    const int lane  = tid & 31;
    const int wv    = tid >> 5;
    const int m     = lane & 15;      // M row (A) / N col (B) owned by lane
    const int hi    = lane >> 4;      // K-half select per ISA 16-bit layout
    const int r     = wv >> 2;        // output row within block (0 or 1)
    const int xbase = (wv & 3) * 32;  // two M-tiles: xbase, xbase+16

    // Per-lane fragment bases. Lane's K pairs = {2j}+8*hi and {16+2j}+8*hi:
    // two aligned uint4 loads per fragment.
    const __bf16* aBase  = patch + (xbase + m) * CH_ + hi * 8 + r * ROWSTRIDE;
    const __bf16* b0Base = wlds + m * KDIM + hi * 8;
    const __bf16* b1Base = wlds + (m + 16) * KDIM + hi * 8;

    union Frag { uint4 q[2]; v16bf v; };
    v8f acc00 = {}, acc01 = {}, acc10 = {}, acc11 = {};

#pragma unroll
    for (int t = 0; t < NTAP; ++t) {
        const int dz = t / 9, dy = (t % 9) / 3, dx = t % 3;
        const int po = ((dz * 4 + dy) * 130 + dx) * CH_;   // compile-time
        Frag a0, a1, bb0, bb1;
        a0.q[0]  = *(const uint4*)(aBase + po);
        a0.q[1]  = *(const uint4*)(aBase + po + 16);
        a1.q[0]  = *(const uint4*)(aBase + po + 16 * CH_);
        a1.q[1]  = *(const uint4*)(aBase + po + 16 * CH_ + 16);
        bb0.q[0] = *(const uint4*)(b0Base + t * CH_);
        bb0.q[1] = *(const uint4*)(b0Base + t * CH_ + 16);
        bb1.q[0] = *(const uint4*)(b1Base + t * CH_);
        bb1.q[1] = *(const uint4*)(b1Base + t * CH_ + 16);
        acc00 = __builtin_amdgcn_wmma_f32_16x16x32_bf16(false, a0.v, false, bb0.v,
                                                        (short)0, acc00, false, false);
        acc01 = __builtin_amdgcn_wmma_f32_16x16x32_bf16(false, a0.v, false, bb1.v,
                                                        (short)0, acc01, false, false);
        acc10 = __builtin_amdgcn_wmma_f32_16x16x32_bf16(false, a1.v, false, bb0.v,
                                                        (short)0, acc10, false, false);
        acc11 = __builtin_amdgcn_wmma_f32_16x16x32_bf16(false, a1.v, false, bb1.v,
                                                        (short)0, acc11, false, false);
    }

    __syncthreads();   // done reading patch; reuse LDS for C staging
    {
        const int vb = r * 128 + xbase;
#pragma unroll
        for (int i = 0; i < 8; ++i) {
            const int mrow = hi * 8 + i;  // C layout: lanes 0-15 M=i, 16-31 M=8+i
            cstage[(vb + mrow) * 32 + m]           = acc00[i];
            cstage[(vb + mrow) * 32 + 16 + m]      = acc01[i];
            cstage[(vb + 16 + mrow) * 32 + m]      = acc10[i];
            cstage[(vb + 16 + mrow) * 32 + 16 + m] = acc11[i];
        }
    }
    __syncthreads();

    // fused CSPN normalization: one thread per voxel (256 voxels = 2 rows)
    {
        float g[27];
#pragma unroll
        for (int ch = 0; ch < 27; ++ch) g[ch] = cstage[tid * 32 + ch];
        float s = 0.f;
#pragma unroll
        for (int i = 0; i < 26; ++i) s += fabsf(g[i]);
        const float inv = 1.0f / s;
        const size_t p = (size_t)bid * 256 + tid;
        float wsum = 0.f;
#pragma unroll
        for (int i = 0; i < 26; ++i) {
            const float wi = g[i] * inv;
            wsum += wi;
            wpl[(size_t)i * NP + p] = (__bf16)wi;
        }
        wpl[(size_t)26 * NP + p] = (__bf16)(1.0f - wsum);
        x0[p] = g[26];   // blurred volume v = conv(x, w_c)
    }
}

// ---------------------------------------------------------------------------
// One CSPN propagation round: x' = w0*x + sum_i w_i * x[neighbor_i].
// Bandwidth-bound; working set (~31 MB) is L2-resident on MI455X (192 MB).
// ---------------------------------------------------------------------------
__global__ __launch_bounds__(256) void cspn_round(const __bf16* __restrict__ wpl,
                                                  const float* __restrict__ xin,
                                                  float* __restrict__ xout) {
    const int idx = blockIdx.x * 256 + threadIdx.x;          // < NP
    const int x = idx & (W_ - 1);
    const int y = (idx >> 7) & (H_ - 1);
    const int z = (idx >> 13) & (D_ - 1);

    float acc = (float)wpl[(size_t)26 * NP + idx] * xin[idx];
    int i = 0;
#pragma unroll
    for (int dz = -1; dz <= 1; ++dz)
#pragma unroll
        for (int dy = -1; dy <= 1; ++dy)
#pragma unroll
            for (int dx = -1; dx <= 1; ++dx) {
                if (dz == 0 && dy == 0 && dx == 0) continue;
                const bool ok = ((unsigned)(z + dz) < D_) &&
                                ((unsigned)(y + dy) < H_) &&
                                ((unsigned)(x + dx) < W_);
                const int off = (dz * H_ + dy) * W_ + dx;
                const float xv = ok ? xin[idx + off] : 0.f;
                acc = fmaf((float)wpl[(size_t)i * NP + idx], xv, acc);
                ++i;
            }
    xout[idx] = acc;
}

// ---------------------------------------------------------------------------
extern "C" void kernel_launch(void* const* d_in, const int* in_sizes, int n_in,
                              void* d_out, int out_size, void* d_ws, size_t ws_size,
                              hipStream_t stream) {
    const float* x  = (const float*)d_in[0];   // [2,32,32,64,128]
    const float* wg = (const float*)d_in[1];   // [26,32,3,3,3]
    const float* wc = (const float*)d_in[2];   // [1,32,3,3,3]
    float* out = (float*)d_out;                // [2,1,32,64,128]

    char* ws = (char*)d_ws;
    __bf16* wmat = (__bf16*)(ws + WS_WMAT);
    __bf16* wpl  = (__bf16*)(ws + WS_WPL);
    float*  xb0  = (float*)(ws + WS_X0);
    float*  xb1  = (float*)(ws + WS_X1);

    prep_weights<<<32, 256, 0, stream>>>(wg, wc, wmat);
    conv_wmma<<<B_ * D_ * (H_ / 2), 256, 0, stream>>>(x, wmat, wpl, xb0);

    float* bufs[2] = {xb0, xb1};
    for (int r = 0; r < 8; ++r) {
        const float* in = bufs[r & 1];
        float* o = (r == 7) ? out : bufs[(r + 1) & 1];
        cspn_round<<<NP / 256, 256, 0, stream>>>(wpl, in, o);
    }
}
